// ReservoirSTDPReadout_9698036154704
// MI455X (gfx1250) — compile-verified, matrix operand order checked by reading, not run on hardware
//
#include <hip/hip_runtime.h>

// ---------------------------------------------------------------------------
// Reservoir LIF network, MI455X (gfx1250) WMMA implementation.
//   x      : [B=64, C=1, T=128]  f32
//   W_in   : [N=4096, 1]         f32
//   W_res  : [N=4096, N=4096]    f32  -> split into f16 hi + f16 lo once
//   W_out  : [10, N=4096]        f32
//   out    : [B=64, 10]          f32  spike counts
//
// Per step: cur = x_t*W_in + s @ W_res^T ; LIF update ; readout LIF.
// GEMM: v_wmma_f32_16x16x32_f16, fp32 accum, hi/lo-split weights (spikes are
// exactly {0,1} in f16, so A*B products are exact; hi+lo gives ~22 mantissa
// bits on the weights).
// Spike matrix (A) is staged per-block into LDS with the gfx1250 async path
// (global_load_async_to_lds_b128 + s_wait_asynccnt), double-buffered, so the
// dominant L2 traffic is just the 64 MB weight stream per step.
// ---------------------------------------------------------------------------

#define BETA   0.9f
#define THRESH 1.0f
#define BATCH  64
#define NRES   4096
#define TSTEPS 128
#define NCLS   10

#define KCH        128                 // K-chunk staged in LDS
#define NCHUNK     (NRES / KCH)        // 32
#define A_STRIDE   (KCH + 8)           // padded row stride (elems) -> 272 B
#define SEGS_PER_ROW (KCH / 8)         // 16-byte segments per row
#define SEGS_PER_CHUNK (BATCH * SEGS_PER_ROW)  // 1024

typedef __attribute__((ext_vector_type(16))) _Float16 v16h;
typedef __attribute__((ext_vector_type(8)))  float    v8f;

union Frag16 {            // one 16x32 f16 WMMA operand = 8 VGPRs/lane
    v16h  h;
    uint4 q[2];
};

// ---------------- W_res f32 -> (hi,lo) f16 split ---------------------------
__global__ __launch_bounds__(256)
void convert_w_kernel(const float* __restrict__ w,
                      _Float16* __restrict__ whi,
                      _Float16* __restrict__ wlo, int n) {
    int i = blockIdx.x * blockDim.x + threadIdx.x;
    if (i < n) {
        float v = w[i];
        _Float16 hi = (_Float16)v;
        _Float16 lo = (_Float16)(v - (float)hi);
        whi[i] = hi;
        wlo[i] = lo;
    }
}

// ---------------- zero persistent state + output ---------------------------
__global__ __launch_bounds__(256)
void init_state_kernel(uint4* __restrict__ state, size_t n16,
                       float* __restrict__ out, int nout) {
    size_t i = (size_t)blockIdx.x * blockDim.x + threadIdx.x;
    if (i < n16) state[i] = make_uint4(0u, 0u, 0u, 0u);
    if (i < (size_t)nout) out[i] = 0.0f;
}

// ---------------- async 16B copy global -> LDS (ASYNCcnt tracked) ----------
__device__ __forceinline__ void async_copy_b128(const _Float16* lds_dst,
                                                const _Float16* gsrc) {
    unsigned int lds_off = (unsigned int)(uintptr_t)lds_dst;  // low 32b = LDS offset
    unsigned long long gaddr = (unsigned long long)(uintptr_t)gsrc;
    asm volatile("global_load_async_to_lds_b128 %0, %1, off"
                 :: "v"(lds_off), "v"(gaddr)
                 : "memory");
}

__device__ __forceinline__ void wait_asynccnt0() {
    asm volatile("s_wait_asynccnt 0x0" ::: "memory");
}

// ---------------- one reservoir timestep -----------------------------------
// Grid: 32 blocks x 256 threads (8 waves). Each wave owns a 16-neuron column
// strip (block covers 128 columns) and the full batch M=64 as four 16-row
// tiles. The block cooperatively stages s_prev K-chunks into LDS.
__global__ __launch_bounds__(256)
void reservoir_step_kernel(const _Float16* __restrict__ s_prev,  // [B, N] f16
                           const _Float16* __restrict__ whi,     // [N, N] f16
                           const _Float16* __restrict__ wlo,     // [N, N] f16
                           const float*    __restrict__ x,       // [B, 1, T]
                           const float*    __restrict__ w_in,    // [N, 1]
                           float*          __restrict__ v_r,     // [B, N]
                           _Float16*       __restrict__ s_new,   // [B, N]
                           int t) {
    __shared__ __align__(16) _Float16 smA[2][BATCH * A_STRIDE];

    const int wib  = threadIdx.x >> 5;               // wave in block (0..7)
    const int lane = threadIdx.x & 31;
    const int hgrp = lane >> 4;                      // lane group (K/M halves)
    const int lrow = lane & 15;
    const int ncol = (blockIdx.x * 8 + wib) * 16 + lrow;  // neuron column

    // B-fragment bases: B[k][n] = W[n][k]; column `ncol` is W row ncol.
    const _Float16* bhi = whi + (size_t)ncol * NRES + hgrp * 8;
    const _Float16* blo = wlo + (size_t)ncol * NRES + hgrp * 8;

    v8f cc[4] = {v8f{}, v8f{}, v8f{}, v8f{}};

    // ---- stage chunk 0 ----
    for (int s = threadIdx.x; s < SEGS_PER_CHUNK; s += 256) {
        int row = s >> 4;            // SEGS_PER_ROW == 16
        int seg = s & 15;
        async_copy_b128(&smA[0][row * A_STRIDE + seg * 8],
                        s_prev + (size_t)row * NRES + seg * 8);
    }

    for (int c = 0; c < NCHUNK; ++c) {
        wait_asynccnt0();            // this wave's staged copies are in LDS
        __syncthreads();             // whole block's copies visible; prev buf free

        if (c + 1 < NCHUNK) {        // stage next chunk into other buffer
            const _Float16* src = s_prev + (size_t)(c + 1) * KCH;
            _Float16* dst = smA[(c + 1) & 1];
            for (int s = threadIdx.x; s < SEGS_PER_CHUNK; s += 256) {
                int row = s >> 4;
                int seg = s & 15;
                async_copy_b128(&dst[row * A_STRIDE + seg * 8],
                                src + (size_t)row * NRES + seg * 8);
            }
        }

        const int buf = c & 1;
#pragma unroll
        for (int ki = 0; ki < KCH / 32; ++ki) {
            const int k0 = c * KCH + ki * 32;        // global K (for B)
            const int kl = ki * 32;                  // chunk-local K (for A)

            // WGP-scope prefetch: pulls upcoming weight lines into all cache
            // levels (locality 3 -> near-scope), not just GL2.
            __builtin_prefetch((const void*)(bhi + k0 + 1024), 0, 3);
            __builtin_prefetch((const void*)(blo + k0 + 1024), 0, 3);

            // B fragments (weights hi/lo) from global (served by L2/WGP$)
            Frag16 bh, bl;
            bh.q[0] = *(const uint4*)(bhi + k0);
            bh.q[1] = *(const uint4*)(bhi + k0 + 16);
            bl.q[0] = *(const uint4*)(blo + k0);
            bl.q[1] = *(const uint4*)(blo + k0 + 16);

            // A fragments (spikes) for all four M-tiles, loaded up-front so
            // the ds_load_b128s issue as one clause and overlap the WMMAs.
            Frag16 a[4];
#pragma unroll
            for (int mt = 0; mt < 4; ++mt) {
                const _Float16* ap =
                    &smA[buf][(mt * 16 + lrow) * A_STRIDE + kl + hgrp * 8];
                a[mt].q[0] = *(const uint4*)(ap);
                a[mt].q[1] = *(const uint4*)(ap + 16);
            }

#pragma unroll
            for (int mt = 0; mt < 4; ++mt) {
                cc[mt] = __builtin_amdgcn_wmma_f32_16x16x32_f16(
                    false, a[mt].h, false, bh.h, (short)0, cc[mt], false, false);
                cc[mt] = __builtin_amdgcn_wmma_f32_16x16x32_f16(
                    false, a[mt].h, false, bl.h, (short)0, cc[mt], false, false);
            }
        }
        __syncthreads();             // all waves done with smA[buf]
    }

    // Epilogue: C/D layout -> element (m, n): vgpr r = m%8, lane = n + 16*(m/8)
    const float win = w_in[ncol];
#pragma unroll
    for (int mt = 0; mt < 4; ++mt) {
#pragma unroll
        for (int r = 0; r < 8; ++r) {
            int brow = mt * 16 + hgrp * 8 + r;               // batch index
            float cur = cc[mt][r] + x[brow * TSTEPS + t] * win;
            size_t idx = (size_t)brow * NRES + ncol;
            float v = BETA * v_r[idx] + cur;
            float sp = (v >= THRESH) ? 1.0f : 0.0f;
            v_r[idx] = v * (1.0f - sp);
            s_new[idx] = (_Float16)sp;
        }
    }
}

// ---------------- readout LIF + spike count --------------------------------
__global__ __launch_bounds__(256)
void readout_step_kernel(const _Float16* __restrict__ s_new,   // [B, N]
                         const float*    __restrict__ w_out,   // [10, N]
                         float*          __restrict__ v_c,     // [B, 10]
                         float*          __restrict__ counts) { // [B, 10]
    __shared__ float red[256];
    const int b   = blockIdx.x;
    const int tid = threadIdx.x;

    float acc[NCLS];
#pragma unroll
    for (int j = 0; j < NCLS; ++j) acc[j] = 0.0f;

    for (int k = tid; k < NRES; k += 256) {
        float sv = (float)s_new[(size_t)b * NRES + k];
#pragma unroll
        for (int j = 0; j < NCLS; ++j)
            acc[j] += sv * w_out[j * NRES + k];
    }

#pragma unroll
    for (int j = 0; j < NCLS; ++j) {
        red[tid] = acc[j];
        __syncthreads();
        for (int off = 128; off > 0; off >>= 1) {
            if (tid < off) red[tid] += red[tid + off];
            __syncthreads();
        }
        if (tid == 0) {
            float v  = BETA * v_c[b * NCLS + j] + red[0];
            float sp = (v >= THRESH) ? 1.0f : 0.0f;
            v_c[b * NCLS + j] = v * (1.0f - sp);
            counts[b * NCLS + j] += sp;
        }
        __syncthreads();
    }
}

// ---------------------------------------------------------------------------
extern "C" void kernel_launch(void* const* d_in, const int* in_sizes, int n_in,
                              void* d_out, int out_size, void* d_ws, size_t ws_size,
                              hipStream_t stream) {
    const float* x     = (const float*)d_in[0];   // [64,1,128]
    const float* w_in  = (const float*)d_in[1];   // [4096,1]
    const float* w_res = (const float*)d_in[2];   // [4096,4096]
    const float* w_out = (const float*)d_in[3];   // [10,4096]
    float*       out   = (float*)d_out;           // [64,10]

    // workspace layout
    char* ws = (char*)d_ws;
    _Float16* whi = (_Float16*)ws;                                   // 32 MB
    _Float16* wlo = (_Float16*)(ws + (size_t)NRES * NRES * 2);       // 32 MB
    char* state = ws + (size_t)NRES * NRES * 4;
    float*    v_r = (float*)state;                                   // 1 MB
    _Float16* s0  = (_Float16*)(state + (size_t)BATCH * NRES * 4);   // 512 KB
    _Float16* s1  = (_Float16*)((char*)s0 + (size_t)BATCH * NRES * 2);
    float*    v_c = (float*)((char*)s1 + (size_t)BATCH * NRES * 2);  // 2.5 KB
    const size_t state_bytes = (size_t)BATCH * NRES * 4
                             + 2 * (size_t)BATCH * NRES * 2
                             + (size_t)BATCH * NCLS * 4;

    // 1) split W_res into f16 hi/lo (idempotent, re-done each launch)
    const int nW = NRES * NRES;
    convert_w_kernel<<<(nW + 255) / 256, 256, 0, stream>>>(w_res, whi, wlo, nW);

    // 2) zero state + output
    const size_t n16 = state_bytes / 16;
    init_state_kernel<<<(int)((n16 + 255) / 256), 256, 0, stream>>>(
        (uint4*)state, n16, out, BATCH * NCLS);

    // 3) 128 sequential timesteps (graph-captured by harness)
    const int resv_blocks = NRES / (8 * 16);      // 32 blocks of 8 waves
    for (int t = 0; t < TSTEPS; ++t) {
        const _Float16* sp = (t & 1) ? s1 : s0;
        _Float16*       sn = (t & 1) ? s0 : s1;
        reservoir_step_kernel<<<resv_blocks, 256, 0, stream>>>(
            sp, whi, wlo, x, w_in, v_r, sn, t);
        readout_step_kernel<<<BATCH, 256, 0, stream>>>(sn, w_out, v_c, out);
    }
    (void)in_sizes; (void)n_in; (void)out_size; (void)ws_size;
}
